// TopologyHead_87136296501639
// MI455X (gfx1250) — matrix operand order repaired and checked by reading, not compiled, not run on hardware
//
#include <hip/hip_runtime.h>
#include <hip/hip_bf16.h>

typedef __bf16 v16bf __attribute__((ext_vector_type(16)));
typedef float  v8f   __attribute__((ext_vector_type(8)));

#define ACT_RELU    0
#define ACT_SIGMOID 1
#define ACT_NONE    2

__device__ __forceinline__ float sigmoidf_(float x) {
    return 1.0f / (1.0f + __expf(-x));
}

__device__ __forceinline__ void pack4_bf16(v16bf& dst, int base, float4 q) {
    dst[base + 0] = (__bf16)q.x;  dst[base + 1] = (__bf16)q.y;
    dst[base + 2] = (__bf16)q.z;  dst[base + 3] = (__bf16)q.w;
}

// ---------------------------------------------------------------------------
// WMMA GEMM:  D[m,n] = act( sum_k A[m,k] * W[n,k] + bias[n] )
// A: (M x K) fp32 row-major, W: (N x K) fp32 row-major. All strides are
// compile-time so the k-loop fully unrolls into back-to-back
// v_wmma_f32_16x16x32_bf16 ops and addressing folds into immediate offsets.
// One wave computes one 16x16 D tile; blockDim=(32,4): each wave owns an
// N-tile. Fragment layouts per CDNA5 ISA 7.12.2 (wave32):
//   A (16x32 bf16): lane=hi*16+m; VGPR0-3: K=hi*8+{0..7}; VGPR4-7: K=16+hi*8+{0..7}
//   B (32x16 bf16): lane=hi*16+n; VGPR0-3: K=hi*8+{0..7}; VGPR4-7: K=16+hi*8+{0..7}
//   D (16x16 f32) : lane=hi*16+n; VGPR v: row hi*8+v
// ---------------------------------------------------------------------------
template <int ACT, int K, int LDA, int LDW, int LDD>
__global__ __launch_bounds__(128)
void wmma_gemm_bias_act(const float* __restrict__ A,
                        const float* __restrict__ W,
                        const float* __restrict__ bias,
                        float* __restrict__ D)
{
    const int lane = threadIdx.x;          // 0..31
    const int wave = threadIdx.y;          // 0..3
    const int m0   = blockIdx.x * 16;
    const int n0   = (blockIdx.y * 4 + wave) * 16;
    const int half = lane >> 4;            // 0 or 1
    const int lid  = lane & 15;

    const float* __restrict__ Arow = A + (long)(m0 + lid) * LDA + half * 8;
    const float* __restrict__ Wrow = W + (long)(n0 + lid) * LDW + half * 8;

    v8f c = {};

    #pragma unroll
    for (int kb = 0; kb < K; kb += 32) {
        v16bf a, b;
        // A: 16 floats per lane = four contiguous float4 runs -> global_load_b128
        pack4_bf16(a, 0,  *(const float4*)(Arow + kb));
        pack4_bf16(a, 4,  *(const float4*)(Arow + kb + 4));
        pack4_bf16(a, 8,  *(const float4*)(Arow + kb + 16));
        pack4_bf16(a, 12, *(const float4*)(Arow + kb + 20));
        // B: same K order as A: four float4 runs -> four global_load_b128
        pack4_bf16(b, 0,  *(const float4*)(Wrow + kb));
        pack4_bf16(b, 4,  *(const float4*)(Wrow + kb + 4));
        pack4_bf16(b, 8,  *(const float4*)(Wrow + kb + 16));
        pack4_bf16(b, 12, *(const float4*)(Wrow + kb + 20));

        c = __builtin_amdgcn_wmma_f32_16x16x32_bf16(
                /*neg_a=*/false, a, /*neg_b=*/false, b,
                /*c_mod=*/(short)0, c, /*reuse_a=*/false, /*reuse_b=*/false);
    }

    // Epilogue: lane -> column n = n0+lid; VGPR v -> row m0 + half*8 + v.
    // Stores across lanes 0..15 hit 16 consecutive n -> coalesced 64B.
    const int   n  = n0 + lid;
    const float bn = bias[n];
    float* __restrict__ Drow = D + (long)(m0 + half * 8) * LDD + n;
    #pragma unroll
    for (int v = 0; v < 8; ++v) {
        float x = c[v] + bn;
        if (ACT == ACT_RELU)         x = fmaxf(x, 0.0f);
        else if (ACT == ACT_SIGMOID) x = sigmoidf_(x);
        Drow[v * LDD] = x;
    }
}

// ---------------------------------------------------------------------------
// Prep: Wstack (128 x 64): rows 0-63 = Wc1[n, 0:64] (W1a),
//                          rows 64-127 = Wc1[n-64, 64:128] (W1b).
// biasP (128): [bc1 | 0] so GEMM3 folds bc1 into p1 only.
// ---------------------------------------------------------------------------
__global__ void prep_kernel(const float* __restrict__ Wc1,
                            const float* __restrict__ bc1,
                            float* __restrict__ Wstack,
                            float* __restrict__ biasP)
{
    int t = blockIdx.x * blockDim.x + threadIdx.x;
    if (t < 128 * 64) {
        int n = t >> 6, d = t & 63;
        Wstack[t] = (n < 64) ? Wc1[n * 128 + d] : Wc1[(n - 64) * 128 + 64 + d];
    }
    if (t < 128) biasP[t] = (t < 64) ? bc1[t] : 0.0f;
}

// ---------------------------------------------------------------------------
// Pairwise head: out[b,i,j] = sigmoid(bc2 + sum_k relu(P[b,i,k] + P[b,j,64+k]) * w2[k])
// P is (1536 x 128): cols 0-63 = p1 + bc1, cols 64-127 = p2.
// 16x16 (i,j) tile per 256-thread block; LDS row stride 68 floats (272B = 17x16B)
// -> 16B-aligned float4 rows, conflict-free banks (4j+k distinct mod 64).
// ---------------------------------------------------------------------------
__global__ __launch_bounds__(256)
void pairwise_kernel(const float* __restrict__ P,
                     const float* __restrict__ Wc2,
                     const float* __restrict__ bc2,
                     float* __restrict__ out)
{
    __shared__ float s1[16 * 68];
    __shared__ float s2[16 * 68];
    __shared__ float sw[64];

    const int b  = blockIdx.z;
    const int i0 = blockIdx.x * 16;
    const int j0 = blockIdx.y * 16;
    const int t  = threadIdx.x;
    const long rowBase = (long)b * 768;

    // 256 threads load 16 rows x 64 cols of each half as float4: 256 b128 loads
    {
        int row = t >> 4, k4 = (t & 15) * 4;        // k4 in {0,4,...,60}
        *(float4*)&s1[row * 68 + k4] = *(const float4*)&P[(rowBase + i0 + row) * 128 + k4];
        *(float4*)&s2[row * 68 + k4] = *(const float4*)&P[(rowBase + j0 + row) * 128 + 64 + k4];
    }
    if (t < 64) sw[t] = Wc2[t];
    __syncthreads();

    const int il = t >> 4, jl = t & 15;
    const float4* __restrict__ r1 = (const float4*)&s1[il * 68];
    const float4* __restrict__ r2 = (const float4*)&s2[jl * 68];
    const float4* __restrict__ w4 = (const float4*)sw;

    float acc = 0.0f;
    #pragma unroll
    for (int k4 = 0; k4 < 16; ++k4) {
        float4 a = r1[k4], bb = r2[k4], w = w4[k4];
        acc += fmaxf(a.x + bb.x, 0.0f) * w.x;
        acc += fmaxf(a.y + bb.y, 0.0f) * w.y;
        acc += fmaxf(a.z + bb.z, 0.0f) * w.z;
        acc += fmaxf(a.w + bb.w, 0.0f) * w.w;
    }
    out[(rowBase + i0 + il) * 768 + (j0 + jl)] = sigmoidf_(acc + bc2[0]);
}

// ---------------------------------------------------------------------------
// Launch
// ---------------------------------------------------------------------------
extern "C" void kernel_launch(void* const* d_in, const int* in_sizes, int n_in,
                              void* d_out, int out_size, void* d_ws, size_t ws_size,
                              hipStream_t stream)
{
    const float* vertex = (const float*)d_in[0];   // (2,768,256)
    const float* We1    = (const float*)d_in[1];   // (128,256)
    const float* be1    = (const float*)d_in[2];   // (128)
    const float* We2    = (const float*)d_in[3];   // (64,128)
    const float* be2    = (const float*)d_in[4];   // (64)
    const float* Wc1    = (const float*)d_in[5];   // (64,128)
    const float* bc1    = (const float*)d_in[6];   // (64)
    const float* Wc2    = (const float*)d_in[7];   // (1,64)
    const float* bc2    = (const float*)d_in[8];   // (1)
    float* out = (float*)d_out;                    // (2,768,768)

    char* ws = (char*)d_ws;
    float* h      = (float*)(ws);                  // 1536*128*4 = 786432 B
    float* e      = (float*)(ws + 786432);         // 1536*64*4  = 393216 B
    float* P      = h;                             // reuse h (disjoint from e)
    float* Wstack = (float*)(ws + 1179648);        // 128*64*4   = 32768 B
    float* biasP  = (float*)(ws + 1212416);        // 128*4      = 512 B

    prep_kernel<<<32, 256, 0, stream>>>(Wc1, bc1, Wstack, biasP);

    dim3 blk(32, 4);
    // GEMM1: h = relu(vertex * We1^T + be1)   M=1536 N=128 K=256
    wmma_gemm_bias_act<ACT_RELU, 256, 256, 256, 128>
        <<<dim3(96, 2), blk, 0, stream>>>(vertex, We1, be1, h);
    // GEMM2: e = sigmoid(h * We2^T + be2)     M=1536 N=64  K=128
    wmma_gemm_bias_act<ACT_SIGMOID, 128, 128, 128, 64>
        <<<dim3(96, 1), blk, 0, stream>>>(h, We2, be2, e);
    // GEMM3: P = e * Wstack^T + [bc1|0]       M=1536 N=128 K=64
    wmma_gemm_bias_act<ACT_NONE, 64, 64, 64, 128>
        <<<dim3(96, 2), blk, 0, stream>>>(e, Wstack, biasP, P);

    // Pairwise relu-reduction + sigmoid
    pairwise_kernel<<<dim3(48, 48, 2), 256, 0, stream>>>(P, Wc2, bc2, out);
}
